// IdentifyLoss_841813590254
// MI455X (gfx1250) — compile-verified, matrix operand order checked by reading, not run on hardware
//
#include <hip/hip_runtime.h>

#define NROWS 8192
#define DIM   128
#define PNUM  4096
#define MACRO 128
#define NT    (NROWS / MACRO)        /* 64  */
#define NBLK  (NT * (NT + 1) / 2)    /* 2080 */
#define BPITCH 136                   /* row pitch in ushorts: 128 + 8 pad (4 dwords) */

typedef __attribute__((ext_vector_type(16))) __bf16        v16bf;
typedef __attribute__((ext_vector_type(16))) unsigned short v16u;
typedef __attribute__((ext_vector_type(8)))  float          v8f;
typedef unsigned long long u64;

// ---- packed sortable (value, index) helpers: fully branchless top-2 ----
__device__ __forceinline__ u64 pack_vi(float v, int idx) {
  unsigned b = __builtin_bit_cast(unsigned, v);
  // order-preserving fp32 -> u32: b>=0 -> b|0x80000000 ; b<0 -> ~b  (no ternary)
  unsigned s = b ^ ((unsigned)((int)b >> 31) | 0x80000000u);
  return ((u64)s << 32) | (unsigned)idx;
}
__device__ __forceinline__ float unpack_v(u64 p) {
  unsigned s = (unsigned)(p >> 32);
  unsigned b = (s & 0x80000000u) ? (s ^ 0x80000000u) : ~s;
  return __builtin_bit_cast(float, b);
}
__device__ __forceinline__ int unpack_i(u64 p) { return (int)(unsigned)(p & 0xffffffffu); }

__device__ __forceinline__ void push2(u64& m1, u64& m2, u64 v) {
  u64 mn = v < m1 ? v : m1;       // v_cmp + v_cndmask, no branches
  m1     = v > m1 ? v : m1;
  m2     = mn > m2 ? mn : m2;
}
__device__ __forceinline__ void merge2(u64& m1, u64& m2, u64 b1, u64 b2) {
  u64 lo = m1 > b1 ? b1 : m1;     // min of the two firsts
  m1     = m1 > b1 ? m1 : b1;
  u64 hi2 = m2 > b2 ? m2 : b2;
  m2     = lo > hi2 ? lo : hi2;
}
__device__ __forceinline__ void wave_reduce2(u64& m1, u64& m2) {
  #pragma unroll
  for (int m = 1; m < 32; m <<= 1) {
    u64 o1 = __shfl_xor(m1, m, 32);
    u64 o2 = __shfl_xor(m2, m, 32);
    merge2(m1, m2, o1, o2);
  }
}

// hi = truncate-to-bf16(v), lo = truncate-to-bf16(v - hi) (compensated split)
__device__ __forceinline__ void split1(float v, unsigned short& h, unsigned short& l) {
  unsigned u = __builtin_bit_cast(unsigned, v);
  h = (unsigned short)(u >> 16);
  float hf = __builtin_bit_cast(float, (unsigned)h << 16);
  unsigned r = __builtin_bit_cast(unsigned, v - hf);
  l = (unsigned short)(r >> 16);
}

// Build one (hi, lo) WMMA A/B fragment from a global fp32 row.
// Fragment striping (16-bit, 16x32): elems 0..7 = K k0+kg.., elems 8..15 = K k0+16+kg.
__device__ __forceinline__ void load_split_frag(const float* __restrict__ rowp,
                                                int k0, int kg,
                                                v16bf& hi, v16bf& lo) {
  const float4* p0 = (const float4*)(rowp + k0 + kg);
  const float4* p1 = (const float4*)(rowp + k0 + 16 + kg);
  float4 f0 = p0[0], f1 = p0[1], f2 = p1[0], f3 = p1[1];
  float vals[16] = { f0.x, f0.y, f0.z, f0.w, f1.x, f1.y, f1.z, f1.w,
                     f2.x, f2.y, f2.z, f2.w, f3.x, f3.y, f3.z, f3.w };
  v16u hu, lu;
  #pragma unroll
  for (int t = 0; t < 16; ++t) {
    unsigned short h, l;
    split1(vals[t], h, l);
    hu[t] = h; lu[t] = l;
  }
  hi = __builtin_bit_cast(v16bf, hu);
  lo = __builtin_bit_cast(v16bf, lu);
}

union FragCast { uint4 q[2]; v16bf v; };

// Fused upper-triangle Gram + per-block top-2. One 128x128 macro-tile per WG.
__global__ __launch_bounds__(256) void gram_top2_kernel(const float* __restrict__ x,
                                                        u64* __restrict__ partial) {
  // decode blockIdx -> (bi, bj) with bj >= bi
  int t = blockIdx.x, bi = 0, rowlen = NT;
  while (t >= rowlen) { t -= rowlen; ++bi; --rowlen; }
  const int bj = bi + t;

  const int tid  = threadIdx.x;
  const int lane = tid & 31;
  const int wave = tid >> 5;
  const int kg   = (lane >> 4) * 8;   // 0 or 8
  const int lrow = lane & 15;

  // B halves staged pre-split into LDS (64 rows x 128 K, bf16 hi/lo, padded pitch)
  __shared__ unsigned short sBhi[64 * BPITCH];
  __shared__ unsigned short sBlo[64 * BPITCH];

  // --- A fragments: split once per wave, kept in registers across the tc loop ---
  const int tr = wave;
  const int i0 = bi * MACRO + tr * 16;
  const float* arow = x + (size_t)(i0 + lrow) * DIM;
  __builtin_prefetch(arow, 0, 0);     // global_prefetch_b8
  v16bf ahi[4], alo[4];
  #pragma unroll
  for (int kc = 0; kc < 4; ++kc)
    load_split_frag(arow, kc * 32, kg, ahi[kc], alo[kc]);

  u64 m1 = 0, m2 = 0;   // packed (sortable-value, index); 0 == -inf sentinel

  for (int half = 0; half < 2; ++half) {
    __syncthreads();   // previous half's LDS no longer in use
    // --- cooperative B staging: 64 rows x 128 K, split fp32 -> bf16 hi/lo ---
    const int jbase = bj * MACRO + half * 64;
    for (int it = 0; it < 8; ++it) {
      const int cid = tid + it * 256;        // 2048 float4-chunks
      const int r   = cid >> 5;              // 0..63
      const int c4  = (cid & 31) * 4;        // k offset, multiple of 4
      float4 f = *(const float4*)(x + (size_t)(jbase + r) * DIM + c4);
      unsigned short h0, l0, h1, l1, h2, l2, h3, l3;
      split1(f.x, h0, l0); split1(f.y, h1, l1);
      split1(f.z, h2, l2); split1(f.w, h3, l3);
      uint2 hp, lp;
      hp.x = (unsigned)h0 | ((unsigned)h1 << 16);
      hp.y = (unsigned)h2 | ((unsigned)h3 << 16);
      lp.x = (unsigned)l0 | ((unsigned)l1 << 16);
      lp.y = (unsigned)l2 | ((unsigned)l3 << 16);
      *(uint2*)(&sBhi[r * BPITCH + c4]) = hp;
      *(uint2*)(&sBlo[r * BPITCH + c4]) = lp;
    }
    __syncthreads();

    // --- compute: 4 subtile columns in this half ---
    for (int tcl = 0; tcl < 4; ++tcl) {
      const int tc = half * 4 + tcl;
      if (bi == bj && tc < tr) continue;   // fully below diagonal
      const int j0 = bj * MACRO + tc * 16;
      const int rloc = tc * 16 + lrow - half * 64;   // 0..63 within staged half

      v8f c = {0.f, 0.f, 0.f, 0.f, 0.f, 0.f, 0.f, 0.f};
      #pragma unroll
      for (int kc = 0; kc < 4; ++kc) {
        const int k0 = kc * 32;
        const unsigned short* bh = &sBhi[rloc * BPITCH];
        const unsigned short* bl = &sBlo[rloc * BPITCH];
        FragCast fh, fl;
        fh.q[0] = *(const uint4*)(bh + k0 + kg);        // ds_load_b128
        fh.q[1] = *(const uint4*)(bh + k0 + 16 + kg);
        fl.q[0] = *(const uint4*)(bl + k0 + kg);
        fl.q[1] = *(const uint4*)(bl + k0 + 16 + kg);
        // compensated bf16 product: hi*hi + hi*lo + lo*hi ~ fp32
        c = __builtin_amdgcn_wmma_f32_16x16x32_bf16(false, ahi[kc], false, fh.v, (short)0, c, false, false);
        c = __builtin_amdgcn_wmma_f32_16x16x32_bf16(false, ahi[kc], false, fl.v, (short)0, c, false, false);
        c = __builtin_amdgcn_wmma_f32_16x16x32_bf16(false, alo[kc], false, fh.v, (short)0, c, false, false);
      }

      // C/D layout: VGPR r -> M = r + 8*(lane>=16), N = lane & 15
      const int m_base = (lane >> 4) * 8;
      const int ncol   = lane & 15;
      #pragma unroll
      for (int r = 0; r < 8; ++r) {
        const int gi = i0 + m_base + r;
        const int gj = j0 + ncol;
        // validity as a bitmask AND (no select, no exec regions)
        const u64 vmask = 0ull - (u64)(gi < gj);
        const u64 pv = pack_vi(c[r], gi * NROWS + gj) & vmask;
        push2(m1, m2, pv);         // fully branchless
      }
    }
  }

  wave_reduce2(m1, m2);
  __shared__ u64 sw1[8], sw2[8];
  if (lane == 0) { sw1[wave] = m1; sw2[wave] = m2; }
  __syncthreads();
  if (tid == 0) {
    u64 a1 = sw1[0], a2 = sw2[0];
    for (int w = 1; w < 8; ++w) merge2(a1, a2, sw1[w], sw2[w]);
    partial[2 * blockIdx.x]     = a1;
    partial[2 * blockIdx.x + 1] = a2;
  }
}

// sim_self[p] = dot(x[p], x[p+1]) in fp32
__global__ __launch_bounds__(256) void self_sim_kernel(const float* __restrict__ x,
                                                       float* __restrict__ sim_self) {
  const int p = blockIdx.x * blockDim.x + threadIdx.x;
  if (p >= PNUM) return;
  const float4* a = (const float4*)(x + (size_t)p * DIM);
  const float4* b = (const float4*)(x + (size_t)(p + 1) * DIM);
  float s = 0.f;
  #pragma unroll
  for (int k = 0; k < DIM / 4; ++k) {
    float4 av = a[k], bv = b[k];
    s += av.x * bv.x + av.y * bv.y + av.z * bv.z + av.w * bv.w;
  }
  sim_self[p] = s;
}

// Merge block partials -> global top-2, then mean(sim_oth / sim_self)
__global__ __launch_bounds__(256) void finalize_kernel(const u64* __restrict__ partial,
                                                       const float* __restrict__ sim_self,
                                                       float* __restrict__ out) {
  const int lane = threadIdx.x & 31;
  const int wave = threadIdx.x >> 5;

  u64 m1 = 0, m2 = 0;
  for (int i = threadIdx.x; i < NBLK; i += 256)
    merge2(m1, m2, partial[2 * i], partial[2 * i + 1]);
  wave_reduce2(m1, m2);

  __shared__ u64 sw1[8], sw2[8];
  __shared__ u64 g1, g2;
  if (lane == 0) { sw1[wave] = m1; sw2[wave] = m2; }
  __syncthreads();
  if (threadIdx.x == 0) {
    u64 a1 = sw1[0], a2 = sw2[0];
    for (int w = 1; w < 8; ++w) merge2(a1, a2, sw1[w], sw2[w]);
    g1 = a1; g2 = a2;
  }
  __syncthreads();

  const float v1 = unpack_v(g1);
  const int   i1 = unpack_i(g1);
  const float v2 = unpack_v(g2);

  float sum = 0.f;
  for (int p = threadIdx.x; p < PNUM; p += 256) {
    const int selfflat = p * NROWS + p + 1;            // < 2^26, fits in int
    const float num = (i1 == selfflat) ? v2 : v1;
    sum += num / sim_self[p];
  }
  #pragma unroll
  for (int m = 1; m < 32; m <<= 1) sum += __shfl_xor(sum, m, 32);
  __shared__ float ss[8];
  if (lane == 0) ss[wave] = sum;
  __syncthreads();
  if (threadIdx.x == 0) {
    float tot = 0.f;
    for (int w = 0; w < 8; ++w) tot += ss[w];
    out[0] = tot / (float)PNUM;
  }
}

extern "C" void kernel_launch(void* const* d_in, const int* in_sizes, int n_in,
                              void* d_out, int out_size, void* d_ws, size_t ws_size,
                              hipStream_t stream) {
  const float* x = (const float*)d_in[0];
  u64*   partial  = (u64*)d_ws;
  float* sim_self = (float*)((char*)d_ws + (size_t)NBLK * 2 * sizeof(u64));
  float* out      = (float*)d_out;

  self_sim_kernel<<<PNUM / 256, 256, 0, stream>>>(x, sim_self);
  gram_top2_kernel<<<NBLK, 256, 0, stream>>>(x, partial);
  finalize_kernel<<<1, 256, 0, stream>>>(partial, sim_self, out);
}
